// SimpleSelfAttention_45578192945952
// MI455X (gfx1250) — matrix operand order repaired and checked by reading, not compiled, not run on hardware
//
#include <hip/hip_runtime.h>
#include <stdint.h>
#include <stddef.h>

// ---------------------------------------------------------------------------
// Pre-LN single-head self-attention for MI455X (gfx1250).
//   B=8, S=2048, D=1024.  bf16 WMMA GEMMs, async global->LDS double buffering,
//   single barrier per K-step.
// Workspace (~198 MB):
//   xn_bf16 [16384,1024], Wq/Wk/Wv bf16 [1024,1024], Q/K bf16 [16384,1024],
//   Vt bf16 [8,1024,2048], P bf16 [8,2048,2048]
// ---------------------------------------------------------------------------

#define D_EMBED 1024
#define BATCH   8
#define SEQ     2048
#define NROWS   (BATCH * SEQ) // 16384

typedef __attribute__((ext_vector_type(16))) __bf16 v16bf;
typedef __attribute__((ext_vector_type(8)))  float  v8f;

// fp32 -> bf16 round-to-nearest-even
__device__ __forceinline__ unsigned short f2bf(float f) {
    unsigned int u = __float_as_uint(f);
    u += 0x7FFFu + ((u >> 16) & 1u);
    return (unsigned short)(u >> 16);
}

union FragU { uint4 q[2]; v16bf v; };

__device__ __forceinline__ v16bf ld_frag(const unsigned short* p0,
                                         const unsigned short* p1) {
    FragU u;
    u.q[0] = *reinterpret_cast<const uint4*>(p0);
    u.q[1] = *reinterpret_cast<const uint4*>(p1);
    return u.v;
}

// ---------------------------------------------------------------------------
// Async global->LDS copy of one 16B chunk (CDNA5 GLOBAL_LOAD_ASYNC_TO_LDS,
// tracked by ASYNCcnt). GV addressing mode: 64-bit VGPR address, LDS dest
// address in a VGPR. Generic LDS pointers carry the LDS byte address in their
// low 32 bits (ISA 10.2 aperture scheme), so truncation yields the DS address.
// ---------------------------------------------------------------------------
__device__ __forceinline__ void async_copy_b128(const unsigned short* g,
                                                unsigned short* l) {
    unsigned int lds_addr = (unsigned int)(uintptr_t)l;
    unsigned long long gaddr = (unsigned long long)(uintptr_t)g;
    asm volatile("global_load_async_to_lds_b128 %0, %1, off"
                 :: "v"(lds_addr), "v"(gaddr)
                 : "memory");
}

__device__ __forceinline__ void wait_asynccnt_0() {
    asm volatile("s_wait_asynccnt 0x0" ::: "memory");
}

// ---------------------------------------------------------------------------
// GEMM building blocks: 128x128 block tile, BK=32, 8 waves (256 thr, wave32),
// wave tile 32x64 = 2x4 WMMA 16x16 tiles, bf16 A/B, f32 accumulate.
// LDS rows padded to 40 halves (80B): breaks bank conflicts, keeps 16B chunks
// aligned. Two LDS stages for async double buffering (40 KB total).
// ---------------------------------------------------------------------------

// Issue async loads for a 128x32 bf16 tile (row-major src, leading dim `ld`).
// 2 x b128 per thread = 2 ASYNCcnt increments per wave per tile.
__device__ __forceinline__ void async_tile_nt(
    unsigned short (&dst)[128][40],
    const unsigned short* __restrict__ src, int ld, int tid)
{
#pragma unroll
    for (int p = 0; p < 2; ++p) {
        int r = (tid >> 2) + p * 64;   // 0..127
        int c = (tid & 3) * 8;         // 0,8,16,24
        async_copy_b128(src + (long)r * ld + c, &dst[r][c]);
    }
}

// One BK=32 step of the wave's 2x4 WMMA tile grid.
__device__ __forceinline__ void mma_tile(
    const unsigned short (&As)[128][40],
    const unsigned short (&Bs)[128][40],
    int lane, int waveMoff, int waveNoff,
    v8f (&acc)[2][4])
{
    const int mlo   = lane & 15;
    const int hi    = lane >> 4;     // 0 or 1
    const int abase = hi * 8;        // A fragment K base
    const int kbase = hi * 16;       // unused; B fragments also use abase
    v16bf a[2], b[4];
#pragma unroll
    for (int tm = 0; tm < 2; ++tm) {
        const unsigned short* row = &As[waveMoff + tm * 16 + mlo][0];
        a[tm] = ld_frag(row + abase, row + abase + 16);
    }
#pragma unroll
    for (int tn = 0; tn < 4; ++tn) {
        const unsigned short* row = &Bs[waveNoff + tn * 16 + mlo][0];
        b[tn] = ld_frag(row + abase, row + abase + 16);
    }
#pragma unroll
    for (int tm = 0; tm < 2; ++tm)
#pragma unroll
        for (int tn = 0; tn < 4; ++tn)
            acc[tm][tn] = __builtin_amdgcn_wmma_f32_16x16x32_bf16(
                false, a[tm], false, b[tn], (short)0, acc[tm][tn],
                false, false);
}

// NT GEMM mainloop: async double buffering with a SINGLE barrier per K-step.
// Invariant at top of iteration kt: only tile kt's 4 async ops are in flight.
//   wait(0)  -> own tile-kt data landed in LDS
//   barrier  -> all waves landed tile kt; all waves finished compute kt-1,
//               so buf[(kt+1)&1] (last read in kt-1) is free to refill
//   issue kt+1 loads, then compute kt (loads overlap the 8 WMMAs)
__device__ __forceinline__ void gemm_nt_async(
    const unsigned short* __restrict__ A,
    const unsigned short* __restrict__ B,
    int K,
    unsigned short (&As)[2][128][40],
    unsigned short (&Bs)[2][128][40],
    int tid, int lane, int waveMoff, int waveNoff,
    v8f (&acc)[2][4])
{
    const int KT = K / 32;
    async_tile_nt(As[0], A, K, tid);
    async_tile_nt(Bs[0], B, K, tid);
    for (int kt = 0; kt < KT; ++kt) {
        wait_asynccnt_0();
        __syncthreads();
        const int nxt = kt + 1;
        if (nxt < KT) {
            async_tile_nt(As[nxt & 1], A + nxt * 32, K, tid);
            async_tile_nt(Bs[nxt & 1], B + nxt * 32, K, tid);
        }
        mma_tile(As[kt & 1], Bs[kt & 1], lane, waveMoff, waveNoff, acc);
    }
}

// ---------------------------------------------------------------------------
// Kernel 1: fp32 -> bf16 conversion (weights)
// ---------------------------------------------------------------------------
__global__ void cvt_f32_to_bf16(const float* __restrict__ src,
                                unsigned short* __restrict__ dst, int n)
{
    int i = blockIdx.x * blockDim.x + threadIdx.x;
    if (i < n) dst[i] = f2bf(src[i]);
}

// ---------------------------------------------------------------------------
// Kernel 2: per-row LayerNorm, output bf16. One 256-thread block per row.
// ---------------------------------------------------------------------------
__global__ __launch_bounds__(256)
void layernorm_to_bf16(const float* __restrict__ x,
                       const float* __restrict__ gamma,
                       const float* __restrict__ beta,
                       unsigned short* __restrict__ xnb)
{
    __shared__ float rs[256];
    __shared__ float rq[256];
    const long row = blockIdx.x;
    const float* p = x + row * D_EMBED;
    const int tid = threadIdx.x;

    float v[4];
    float s = 0.f, q = 0.f;
#pragma unroll
    for (int i = 0; i < 4; ++i) {
        v[i] = p[tid + i * 256];
        s += v[i];
        q += v[i] * v[i];
    }
    rs[tid] = s; rq[tid] = q;
    __syncthreads();
    for (int off = 128; off > 0; off >>= 1) {
        if (tid < off) { rs[tid] += rs[tid + off]; rq[tid] += rq[tid + off]; }
        __syncthreads();
    }
    const float mu   = rs[0] * (1.0f / D_EMBED);
    const float var  = rq[0] * (1.0f / D_EMBED) - mu * mu;
    const float rstd = rsqrtf(var + 1e-5f);
#pragma unroll
    for (int i = 0; i < 4; ++i) {
        const int c = tid + i * 256;
        const float y = (v[i] - mu) * rstd * gamma[c] + beta[c];
        xnb[row * D_EMBED + c] = f2bf(y);
    }
}

// ---------------------------------------------------------------------------
// Kernel 3: QKV projection (NT GEMM): C[m,n] = sum_k xn[m,k]*W[n,k] + bias[n].
// grid = (D/128, M/128, 3); z selects Q/K/V. Q,K stored [M,D] bf16;
// V stored TRANSPOSED per batch as Vt[b][d][s] so the PV GEMM is also NT.
// ---------------------------------------------------------------------------
__global__ __launch_bounds__(256)
void qkv_gemm(const unsigned short* __restrict__ xnb,
              const unsigned short* __restrict__ Wqb,
              const unsigned short* __restrict__ Wkb,
              const unsigned short* __restrict__ Wvb,
              const float* __restrict__ bq,
              const float* __restrict__ bk,
              const float* __restrict__ bv,
              unsigned short* __restrict__ Qb,
              unsigned short* __restrict__ Kb,
              unsigned short* __restrict__ Vtb)
{
    __shared__ __align__(16) unsigned short As[2][128][40];
    __shared__ __align__(16) unsigned short Bs[2][128][40];

    const int which = blockIdx.z;
    const unsigned short* W   = (which == 0) ? Wqb : (which == 1) ? Wkb : Wvb;
    const float*          bia = (which == 0) ? bq  : (which == 1) ? bk  : bv;

    const int n0 = blockIdx.x * 128;
    const int m0 = blockIdx.y * 128;
    const int tid = threadIdx.x;
    const int lane = tid & 31;
    const int wid  = tid >> 5;
    const int waveMoff = (wid & 3) * 32;
    const int waveNoff = (wid >> 2) * 64;

    v8f acc[2][4];
#pragma unroll
    for (int tm = 0; tm < 2; ++tm)
#pragma unroll
        for (int tn = 0; tn < 4; ++tn)
            acc[tm][tn] = (v8f){0.f, 0.f, 0.f, 0.f, 0.f, 0.f, 0.f, 0.f};

    gemm_nt_async(xnb + (long)m0 * D_EMBED,
                  W   + (long)n0 * D_EMBED,
                  D_EMBED, As, Bs, tid, lane, waveMoff, waveNoff, acc);

    if (which < 2) {
        unsigned short* Out = (which == 0) ? Qb : Kb;
#pragma unroll
        for (int tm = 0; tm < 2; ++tm)
#pragma unroll
            for (int tn = 0; tn < 4; ++tn)
#pragma unroll
                for (int r = 0; r < 8; ++r) {
                    const int gm = m0 + waveMoff + tm * 16 + r + (lane >> 4) * 8;
                    const int gn = n0 + waveNoff + tn * 16 + (lane & 15);
                    Out[(long)gm * D_EMBED + gn] = f2bf(acc[tm][tn][r] + bia[gn]);
                }
    } else {
        // V: store transposed per batch: Vt[b][gn][s], gm = b*SEQ + s.
#pragma unroll
        for (int tm = 0; tm < 2; ++tm)
#pragma unroll
            for (int tn = 0; tn < 4; ++tn)
#pragma unroll
                for (int r = 0; r < 8; ++r) {
                    const int gm = m0 + waveMoff + tm * 16 + r + (lane >> 4) * 8;
                    const int gn = n0 + waveNoff + tn * 16 + (lane & 15);
                    const int b  = gm >> 11;          // / SEQ
                    const int s  = gm & (SEQ - 1);
                    Vtb[(long)b * D_EMBED * SEQ + (long)gn * SEQ + s] =
                        f2bf(acc[tm][tn][r] + bia[gn]);
                }
    }
}

// ---------------------------------------------------------------------------
// Kernel 4: scores = Q @ K^T * (1/sqrt(D)), per batch (NT GEMM, fp32 out).
// grid = (S/128, S/128, B)
// ---------------------------------------------------------------------------
__global__ __launch_bounds__(256)
void scores_gemm(const unsigned short* __restrict__ Qb,
                 const unsigned short* __restrict__ Kb,
                 float* __restrict__ attn)
{
    __shared__ __align__(16) unsigned short As[2][128][40];
    __shared__ __align__(16) unsigned short Bs[2][128][40];

    const int b = blockIdx.z;
    const unsigned short* A  = Qb + (long)b * SEQ * D_EMBED;
    const unsigned short* Bm = Kb + (long)b * SEQ * D_EMBED;
    float* C = attn + (long)b * SEQ * SEQ;

    const int n0 = blockIdx.x * 128;
    const int m0 = blockIdx.y * 128;
    const int tid = threadIdx.x;
    const int lane = tid & 31;
    const int wid  = tid >> 5;
    const int waveMoff = (wid & 3) * 32;
    const int waveNoff = (wid >> 2) * 64;

    v8f acc[2][4];
#pragma unroll
    for (int tm = 0; tm < 2; ++tm)
#pragma unroll
        for (int tn = 0; tn < 4; ++tn)
            acc[tm][tn] = (v8f){0.f, 0.f, 0.f, 0.f, 0.f, 0.f, 0.f, 0.f};

    gemm_nt_async(A + (long)m0 * D_EMBED,
                  Bm + (long)n0 * D_EMBED,
                  D_EMBED, As, Bs, tid, lane, waveMoff, waveNoff, acc);

    const float scale = 0.03125f; // 1/sqrt(1024)
#pragma unroll
    for (int tm = 0; tm < 2; ++tm)
#pragma unroll
        for (int tn = 0; tn < 4; ++tn)
#pragma unroll
            for (int r = 0; r < 8; ++r) {
                const int gm = m0 + waveMoff + tm * 16 + r + (lane >> 4) * 8;
                const int gn = n0 + waveNoff + tn * 16 + (lane & 15);
                C[(long)gm * SEQ + gn] = acc[tm][tn][r] * scale;
            }
}

// ---------------------------------------------------------------------------
// Kernel 5: row softmax over S=2048: fp32 in place + bf16 copy for PV GEMM.
// ---------------------------------------------------------------------------
__global__ __launch_bounds__(256)
void softmax_rows(float* __restrict__ attn, unsigned short* __restrict__ Pb)
{
    __shared__ float red[256];
    const long row = blockIdx.x;
    float* p = attn + row * SEQ;
    unsigned short* pb = Pb + row * SEQ;
    const int tid = threadIdx.x;

    float x[8];
    float mx = -3.4e38f;
#pragma unroll
    for (int i = 0; i < 8; ++i) {
        x[i] = p[tid + i * 256];
        mx = fmaxf(mx, x[i]);
    }
    red[tid] = mx;
    __syncthreads();
    for (int s = 128; s > 0; s >>= 1) {
        if (tid < s) red[tid] = fmaxf(red[tid], red[tid + s]);
        __syncthreads();
    }
    mx = red[0];
    __syncthreads();

    float sum = 0.f;
#pragma unroll
    for (int i = 0; i < 8; ++i) {
        x[i] = __expf(x[i] - mx);
        sum += x[i];
    }
    red[tid] = sum;
    __syncthreads();
    for (int s = 128; s > 0; s >>= 1) {
        if (tid < s) red[tid] += red[tid + s];
        __syncthreads();
    }
    const float inv = 1.0f / red[0];
#pragma unroll
    for (int i = 0; i < 8; ++i) {
        const float v = x[i] * inv;
        p[tid + i * 256]  = v;
        pb[tid + i * 256] = f2bf(v);
    }
}

// ---------------------------------------------------------------------------
// Kernel 6: out = x + P @ Vt^T per batch (NT GEMM thanks to transposed V).
// grid = (D/128, S/128, B)
// ---------------------------------------------------------------------------
__global__ __launch_bounds__(256)
void av_gemm(const unsigned short* __restrict__ Pb,
             const unsigned short* __restrict__ Vtb,
             const float* __restrict__ x,
             float* __restrict__ out)
{
    __shared__ __align__(16) unsigned short As[2][128][40];
    __shared__ __align__(16) unsigned short Bs[2][128][40];

    const int b = blockIdx.z;
    const unsigned short* A  = Pb  + (long)b * SEQ * SEQ;      // [S, S]
    const unsigned short* Bm = Vtb + (long)b * D_EMBED * SEQ;  // [D, S] (NT)
    const float* R = x   + (long)b * SEQ * D_EMBED;
    float*       C = out + (long)b * SEQ * D_EMBED;

    const int n0 = blockIdx.x * 128;
    const int m0 = blockIdx.y * 128;
    const int tid = threadIdx.x;
    const int lane = tid & 31;
    const int wid  = tid >> 5;
    const int waveMoff = (wid & 3) * 32;
    const int waveNoff = (wid >> 2) * 64;

    v8f acc[2][4];
#pragma unroll
    for (int tm = 0; tm < 2; ++tm)
#pragma unroll
        for (int tn = 0; tn < 4; ++tn)
            acc[tm][tn] = (v8f){0.f, 0.f, 0.f, 0.f, 0.f, 0.f, 0.f, 0.f};

    gemm_nt_async(A  + (long)m0 * SEQ,
                  Bm + (long)n0 * SEQ,
                  SEQ, As, Bs, tid, lane, waveMoff, waveNoff, acc);

#pragma unroll
    for (int tm = 0; tm < 2; ++tm)
#pragma unroll
        for (int tn = 0; tn < 4; ++tn)
#pragma unroll
            for (int r = 0; r < 8; ++r) {
                const int gm = m0 + waveMoff + tm * 16 + r + (lane >> 4) * 8;
                const int gn = n0 + waveNoff + tn * 16 + (lane & 15);
                const long idx = (long)gm * D_EMBED + gn;
                C[idx] = R[idx] + acc[tm][tn][r];
            }
}

// ---------------------------------------------------------------------------
// Host-side launcher
// ---------------------------------------------------------------------------
extern "C" void kernel_launch(void* const* d_in, const int* in_sizes, int n_in,
                              void* d_out, int out_size, void* d_ws, size_t ws_size,
                              hipStream_t stream)
{
    (void)in_sizes; (void)n_in; (void)out_size; (void)ws_size;

    const float* x  = (const float*)d_in[0];
    const float* Wq = (const float*)d_in[1];
    const float* bq = (const float*)d_in[2];
    const float* Wk = (const float*)d_in[3];
    const float* bk = (const float*)d_in[4];
    const float* Wv = (const float*)d_in[5];
    const float* bv = (const float*)d_in[6];
    const float* g  = (const float*)d_in[7];
    const float* be = (const float*)d_in[8];

    float* out  = (float*)d_out;                         // [B,S,D]
    float* attn = out + (size_t)NROWS * D_EMBED;         // [B,S,S]

    char* ws = (char*)d_ws;
    unsigned short* xnb = (unsigned short*)ws; ws += (size_t)NROWS * D_EMBED * 2;
    unsigned short* Wqb = (unsigned short*)ws; ws += (size_t)D_EMBED * D_EMBED * 2;
    unsigned short* Wkb = (unsigned short*)ws; ws += (size_t)D_EMBED * D_EMBED * 2;
    unsigned short* Wvb = (unsigned short*)ws; ws += (size_t)D_EMBED * D_EMBED * 2;
    unsigned short* Qb  = (unsigned short*)ws; ws += (size_t)NROWS * D_EMBED * 2;
    unsigned short* Kb  = (unsigned short*)ws; ws += (size_t)NROWS * D_EMBED * 2;
    unsigned short* Vtb = (unsigned short*)ws; ws += (size_t)NROWS * D_EMBED * 2;
    unsigned short* Pb  = (unsigned short*)ws; ws += (size_t)NROWS * SEQ * 2;

    const int nW = D_EMBED * D_EMBED;
    cvt_f32_to_bf16<<<(nW + 255) / 256, 256, 0, stream>>>(Wq, Wqb, nW);
    cvt_f32_to_bf16<<<(nW + 255) / 256, 256, 0, stream>>>(Wk, Wkb, nW);
    cvt_f32_to_bf16<<<(nW + 255) / 256, 256, 0, stream>>>(Wv, Wvb, nW);

    layernorm_to_bf16<<<NROWS, 256, 0, stream>>>(x, g, be, xnb);

    qkv_gemm<<<dim3(D_EMBED / 128, NROWS / 128, 3), 256, 0, stream>>>(
        xnb, Wqb, Wkb, Wvb, bq, bk, bv, Qb, Kb, Vtb);

    scores_gemm<<<dim3(SEQ / 128, SEQ / 128, BATCH), 256, 0, stream>>>(
        Qb, Kb, attn);

    softmax_rows<<<NROWS, 256, 0, stream>>>(attn, Pb);

    av_gemm<<<dim3(D_EMBED / 128, SEQ / 128, BATCH), 256, 0, stream>>>(
        Pb, Vtb, x, out);
}
